// NonlinearGAT_G_70428873720559
// MI455X (gfx1250) — matrix operand order repaired and checked by reading, not compiled, not run on hardware
//
#include <hip/hip_runtime.h>
#include <math.h>

#define NFEAT   256
#define NHID    64
#define NHEADS  8
#define HID_ALL (NHID * NHEADS)   // 512
#define NCLASS  40
#define NEG_SLOPE 0.2f
#define EPSV 1e-6f

typedef _Float16 half_t;
typedef __attribute__((ext_vector_type(16))) _Float16 v16h;
typedef __attribute__((ext_vector_type(8)))  _Float16 v8h;
typedef __attribute__((ext_vector_type(8)))  float    v8f;
typedef __attribute__((ext_vector_type(4)))  unsigned int v4u;
typedef __attribute__((ext_vector_type(8)))  int          v8i;
typedef __attribute__((ext_vector_type(4)))  int          v4i;

// ---------------------------------------------------------------- helpers
__device__ __forceinline__ float sigmoidf_(float x) { return 1.0f / (1.0f + __expf(-x)); }
__device__ __forceinline__ float leaky_(float x) { return x >= 0.f ? x : NEG_SLOPE * x; }

__device__ __forceinline__ void atomicMaxF(float* addr, float val) {
    int old = __float_as_int(*addr);
    while (__int_as_float(old) < val) {
        int prev = atomicCAS((int*)addr, old, __float_as_int(val));
        if (prev == old) break;
        old = prev;
    }
}
__device__ __forceinline__ void atomicMinF(float* addr, float val) {
    int old = __float_as_int(*addr);
    while (__int_as_float(old) > val) {
        int prev = atomicCAS((int*)addr, old, __float_as_int(val));
        if (prev == old) break;
        old = prev;
    }
}

// ---------------------------------------------------------------- TDM: 2D f16 tile global -> LDS
// D# per CDNA5 ISA 8.3/8.4. group0: count=1|lds_addr|global_addr|type=2.
// group1: data_size=2B, tensor_dim0/1 (OOB reads return zero),
// tile_dim0 x tile_dim1, tensor_dim0_stride. groups 2/3/(+v8i) zero (2D tensor).
// This toolchain exposes the 6-arg builtin: (v4u, v8i, v4i, v4i, v8i, i32 cpol).
__device__ __forceinline__ void tdm_load_tile_f16(
        unsigned lds_off, const void* gaddr,
        unsigned tensor_d0, unsigned tensor_d1,
        unsigned tile_d0, unsigned tile_d1, unsigned stride0) {
    unsigned long long ga = (unsigned long long)gaddr;
    v4u g0;
    g0[0] = 1u;                                            // count=1, user-mode
    g0[1] = lds_off;                                       // LDS byte address
    g0[2] = (unsigned)(ga & 0xFFFFFFFFu);                  // global_addr[31:0]
    g0[3] = (unsigned)((ga >> 32) & 0x01FFFFFFu) | (2u << 30);  // addr[56:32] | type=2
    v8i g1;
    g1[0] = (int)(1u << 16);                               // data_size=1 -> 2 bytes
    g1[1] = (int)((tensor_d0 & 0xFFFFu) << 16);            // tensor_dim0[15:0] @ [63:48]
    g1[2] = (int)((tensor_d0 >> 16) & 0xFFFFu)             // tensor_dim0[31:16]
          | (int)((tensor_d1 & 0xFFFFu) << 16);            // tensor_dim1[15:0]
    g1[3] = (int)((tensor_d1 >> 16) & 0xFFFFu)             // tensor_dim1[31:16]
          | (int)((tile_d0 & 0xFFFFu) << 16);              // tile_dim0 @ [127:112]
    g1[4] = (int)(tile_d1 & 0xFFFFu);                      // tile_dim1 @ [143:128], tile_dim2=0
    g1[5] = (int)stride0;                                  // tensor_dim0_stride[31:0]
    g1[6] = 0;                                             // stride0[47:32], stride1[15:0]
    g1[7] = 0;
    v4i z4 = (v4i){0, 0, 0, 0};
    v8i z8 = (v8i){0, 0, 0, 0, 0, 0, 0, 0};
    __builtin_amdgcn_tensor_load_to_lds(g0, g1, z4, z4, z8, 0);
}

// ---------------------------------------------------------------- fills / converts
__global__ void fill_f32(float* p, float v, long long n) {
    long long i = (long long)blockIdx.x * blockDim.x + threadIdx.x;
    long long stride = (long long)gridDim.x * blockDim.x;
    for (; i < n; i += stride) p[i] = v;
}

__global__ void f32_to_f16(const float* __restrict__ in, half_t* __restrict__ out, long long n) {
    long long i = (long long)blockIdx.x * blockDim.x + threadIdx.x;
    long long stride = (long long)gridDim.x * blockDim.x;
    for (; i < n; i += stride) out[i] = (half_t)in[i];
}

// ---------------------------------------------------------------- WMMA GEMM (TDM-staged f16 B, f16 A)
// FT[Nrows, Ncols] = X[Nrows, Kdim] @ W[Ncols, Kdim]^T   (both K-contiguous, f16)
// Block = 8 waves, wave w -> row tile rowGroup*8+w, each wave does CT col tiles.
// Per k-step, wave 0 issues one TENSOR_LOAD_TO_LDS for the CT*16 x 32 f16 W
// tile (LDS layout lbh[c*32 + k]); a B fragment is one contiguous 32B read.
template <int CT>
__global__ __launch_bounds__(256) void gemm_wmma_tdm(
        const half_t* __restrict__ X, const half_t* __restrict__ W,
        float* __restrict__ FT, int Nrows, int Kdim, int Ncols) {
    __shared__ half_t lbh[CT * 16 * 32];

    const int rowTiles  = (Nrows + 15) >> 4;
    const int colTiles  = (Ncols + 15) >> 4;
    const int colGroups = (colTiles + CT - 1) / CT;
    const int rowGroup  = blockIdx.x / colGroups;
    const int colGroup  = blockIdx.x % colGroups;

    const int waveId = threadIdx.x >> 5;
    const int lane   = threadIdx.x & 31;
    const int m      = lane & 15;
    const int hiA    = (lane >= 16) ? 16 : 0;   // A: lanes 16-31 shift K by +16
    const int hiB    = (lane >= 16) ? 16 : 0;   // B: lanes 16-31 shift K by +16

    const int rowTile = rowGroup * 8 + waveId;
    const bool rowOk  = (rowTile < rowTiles);
    int row = rowTile * 16 + m;
    if (row >= Nrows) row = Nrows - 1;          // clamp: loads stay valid, stores guarded
    const half_t* xr = X + (long long)row * Kdim;

    const int colBase = colGroup * CT * 16;
    const unsigned lds_off = (unsigned)(unsigned long long)(&lbh[0]);

    v8f acc[CT];
#pragma unroll
    for (int ct = 0; ct < CT; ++ct) acc[ct] = (v8f){};

    for (int k0 = 0; k0 < Kdim; k0 += 32) {
        __syncthreads();   // previous iteration's LDS reads done
        if (waveId == 0) {
            const half_t* gsrc = W + (long long)colBase * Kdim + k0;
            tdm_load_tile_f16(lds_off, gsrc,
                              (unsigned)(Kdim - k0),      // tensor_dim0: remaining K
                              (unsigned)(Ncols - colBase),// tensor_dim1: remaining cols (OOB->0)
                              32u, (unsigned)(CT * 16),   // tile 32 x CT*16
                              (unsigned)Kdim);            // row stride (elements)
            __builtin_amdgcn_s_wait_tensorcnt(0);         // per-wave counter
        }
        __syncthreads();   // tile visible to all waves

        // ---- A fragment: two contiguous 8-half runs (K = k0+hiA.., k0+hiA+8..)
        if (k0 + 32 < Kdim) __builtin_prefetch(xr + k0 + 32, 0, 3);
        v8h alo = *(const v8h*)(xr + k0 + hiA);
        v8h ahi = *(const v8h*)(xr + k0 + hiA + 8);
        v16h a;
#pragma unroll
        for (int i = 0; i < 8; ++i) { a[i] = alo[i]; a[8 + i] = ahi[i]; }

        // ---- CT wmma against LDS-resident f16 B fragments (16 contiguous halves)
#pragma unroll
        for (int ct = 0; ct < CT; ++ct) {
            v16h b = *(const v16h*)(lbh + (ct * 16 + m) * 32 + hiB);
            acc[ct] = __builtin_amdgcn_wmma_f32_16x16x32_f16(false, a, false, b,
                                                             (short)0, acc[ct], false, false);
        }
    }

    // ---- store: C layout VGPR r -> M = r + 8*(lane>=16), N = lane%16
    if (!rowOk) return;
    const int baseRow = rowTile * 16 + ((lane >= 16) ? 8 : 0);
#pragma unroll
    for (int ct = 0; ct < CT; ++ct) {
        int ncol = colBase + ct * 16 + m;
        if (ncol >= Ncols) continue;
#pragma unroll
        for (int r = 0; r < 8; ++r) {
            int rr = baseRow + r;
            if (rr < Nrows) FT[(long long)rr * Ncols + ncol] = acc[ct][r];
        }
    }
}

// ---------------------------------------------------------------- per-node dots
__global__ void head_dots(const float* __restrict__ ft, const float* __restrict__ al,
                          const float* __restrict__ ar, float* __restrict__ el,
                          float* __restrict__ er, int N, int nh, int fout) {
    long long t = (long long)blockIdx.x * blockDim.x + threadIdx.x;
    if (t >= (long long)N * nh) return;
    int n = (int)(t / nh), h = (int)(t % nh);
    const float* f = ft + (long long)n * nh * fout + (long long)h * fout;
    const float* a = al + (long long)h * fout;
    const float* b = ar + (long long)h * fout;
    float sl = 0.f, sr = 0.f;
    for (int j = 0; j < fout; ++j) { float v = f[j]; sl += v * a[j]; sr += v * b[j]; }
    el[(long long)n * nh + h] = sl;
    er[(long long)n * nh + h] = sr;
}

// ---------------------------------------------------------------- global min per head
__global__ void min_reduce(const float* __restrict__ ft, float* __restrict__ mu,
                           int N, int stride, int cols) {
    int head = blockIdx.y;
    const float* base = ft + (long long)head * cols;
    long long total = (long long)N * cols;
    float lm = __builtin_inff();
    for (long long i = (long long)blockIdx.x * blockDim.x + threadIdx.x; i < total;
         i += (long long)gridDim.x * blockDim.x) {
        long long n = i / cols;
        int j = (int)(i % cols);
        lm = fminf(lm, base[n * stride + j]);
    }
    __shared__ float sd[256];
    sd[threadIdx.x] = lm;
    __syncthreads();
    for (int s = 128; s > 0; s >>= 1) {
        if (threadIdx.x < s) sd[threadIdx.x] = fminf(sd[threadIdx.x], sd[threadIdx.x + s]);
        __syncthreads();
    }
    if (threadIdx.x == 0) atomicMinF(&mu[head], sd[0]);
}

// ---------------------------------------------------------------- pf = pow(ft - mu + eps, pe)   (in place)
__global__ void pow_fwd(float* __restrict__ ft, const float* __restrict__ mu,
                        const float* __restrict__ p, long long total, int cols, int colsPerHead) {
    long long i = (long long)blockIdx.x * blockDim.x + threadIdx.x;
    long long stride = (long long)gridDim.x * blockDim.x;
    for (; i < total; i += stride) {
        int col = (int)(i % cols);
        int h = col / colsPerHead;
        float pe = sigmoidf_(p[h]) + 1.0f;   // S_F = 1.0
        ft[i] = powf(ft[i] - mu[h] + EPSV, pe);
    }
}

// ---------------------------------------------------------------- edge softmax phases
__global__ void edge_max_k(const int* __restrict__ src, const int* __restrict__ dst,
                           const float* __restrict__ el, const float* __restrict__ er,
                           float* __restrict__ mbuf, int E, int nh) {
    int e = blockIdx.x * blockDim.x + threadIdx.x;
    if (e >= E) return;
    int s = src[e], d = dst[e];
    for (int h = 0; h < nh; ++h)
        atomicMaxF(&mbuf[(long long)d * nh + h],
                   leaky_(el[(long long)s * nh + h] + er[(long long)d * nh + h]));
}

__global__ void edge_sum_k(const int* __restrict__ src, const int* __restrict__ dst,
                           const float* __restrict__ el, const float* __restrict__ er,
                           const float* __restrict__ mbuf, float* __restrict__ sbuf,
                           int E, int nh) {
    int e = blockIdx.x * blockDim.x + threadIdx.x;
    if (e >= E) return;
    int s = src[e], d = dst[e];
    for (int h = 0; h < nh; ++h) {
        float v = leaky_(el[(long long)s * nh + h] + er[(long long)d * nh + h]);
        atomicAdd(&sbuf[(long long)d * nh + h], __expf(v - mbuf[(long long)d * nh + h]));
    }
}

// ---------------------------------------------------------------- aggregation, layer 1 (8 heads x 64 feats)
__global__ void edge_agg1(const int* __restrict__ src, const int* __restrict__ dst,
                          const float* __restrict__ el, const float* __restrict__ er,
                          const float* __restrict__ mbuf, const float* __restrict__ sbuf,
                          const float* __restrict__ pf, float* __restrict__ agg, int E) {
    long long t = (long long)blockIdx.x * blockDim.x + threadIdx.x;
    if (t >= (long long)E * NHEADS) return;
    int e = (int)(t >> 3);
    int h = (int)(t & 7);
    int s = src[e], d = dst[e];
    float v = leaky_(el[(long long)s * NHEADS + h] + er[(long long)d * NHEADS + h]);
    float a = __expf(v - mbuf[(long long)d * NHEADS + h]) / sbuf[(long long)d * NHEADS + h];
    const float4* pp = (const float4*)(pf + (long long)s * HID_ALL + h * NHID);
    float* ap = agg + (long long)d * HID_ALL + h * NHID;
#pragma unroll
    for (int j = 0; j < NHID / 4; ++j) {
        float4 q = pp[j];
        atomicAdd(ap + 4 * j + 0, q.x * a);
        atomicAdd(ap + 4 * j + 1, q.y * a);
        atomicAdd(ap + 4 * j + 2, q.z * a);
        atomicAdd(ap + 4 * j + 3, q.w * a);
    }
}

// ---------------------------------------------------------------- aggregation, layer 2 (1 head x 40 feats)
__global__ void edge_agg2(const int* __restrict__ src, const int* __restrict__ dst,
                          const float* __restrict__ el, const float* __restrict__ er,
                          const float* __restrict__ mbuf, const float* __restrict__ sbuf,
                          const float* __restrict__ pf, float* __restrict__ agg, int E) {
    int e = blockIdx.x * blockDim.x + threadIdx.x;
    if (e >= E) return;
    int s = src[e], d = dst[e];
    float v = leaky_(el[s] + er[d]);
    float a = __expf(v - mbuf[d]) / sbuf[d];
    const float4* pp = (const float4*)(pf + (long long)s * NCLASS);
    float* ap = agg + (long long)d * NCLASS;
#pragma unroll
    for (int j = 0; j < NCLASS / 4; ++j) {
        float4 q = pp[j];
        atomicAdd(ap + 4 * j + 0, q.x * a);
        atomicAdd(ap + 4 * j + 1, q.y * a);
        atomicAdd(ap + 4 * j + 2, q.z * a);
        atomicAdd(ap + 4 * j + 3, q.w * a);
    }
}

// ---------------------------------------------------------------- rst = pow(agg+eps, 1/pe)+mu+b, then elu -> f16 hcat
__global__ void rst_elu1(const float* __restrict__ agg, const float* __restrict__ mu,
                         const float* __restrict__ p, const float* __restrict__ b,
                         half_t* __restrict__ hc, long long total) {
    long long i = (long long)blockIdx.x * blockDim.x + threadIdx.x;
    long long stride = (long long)gridDim.x * blockDim.x;
    for (; i < total; i += stride) {
        int col = (int)(i % HID_ALL);
        int h = col >> 6;
        float pe = sigmoidf_(p[h]) + 1.0f;
        float v = powf(agg[i] + EPSV, 1.0f / pe) + mu[h] + b[col];
        hc[i] = (half_t)(v > 0.f ? v : (__expf(v) - 1.0f));   // elu, f16 for GEMM2
    }
}

// ---------------------------------------------------------------- layer-2 epilogue: inverse pow + bias + log_softmax
__global__ void rst_logsoftmax(const float* __restrict__ agg, const float* __restrict__ mu,
                               const float* __restrict__ p, const float* __restrict__ b,
                               float* __restrict__ out, int N) {
    int n = blockIdx.x * blockDim.x + threadIdx.x;
    if (n >= N) return;
    float pe = sigmoidf_(p[0]) + 1.0f;
    float ip = 1.0f / pe;
    float m0 = mu[0];
    float v[NCLASS];
    float mx = -__builtin_inff();
#pragma unroll
    for (int j = 0; j < NCLASS; ++j) {
        v[j] = powf(agg[(long long)n * NCLASS + j] + EPSV, ip) + m0 + b[j];
        mx = fmaxf(mx, v[j]);
    }
    float se = 0.f;
#pragma unroll
    for (int j = 0; j < NCLASS; ++j) se += __expf(v[j] - mx);
    float l = __logf(se);
#pragma unroll
    for (int j = 0; j < NCLASS; ++j) out[(long long)n * NCLASS + j] = v[j] - mx - l;
}

// ---------------------------------------------------------------- host
static inline int blocks_for(long long n, int bs) { return (int)((n + bs - 1) / bs); }

extern "C" void kernel_launch(void* const* d_in, const int* in_sizes, int n_in,
                              void* d_out, int out_size, void* d_ws, size_t ws_size,
                              hipStream_t stream) {
    const float* x   = (const float*)d_in[0];
    const int*   src = (const int*)d_in[1];
    const int*   dst = (const int*)d_in[2];
    const float* W1  = (const float*)d_in[3];
    const float* al1 = (const float*)d_in[4];
    const float* ar1 = (const float*)d_in[5];
    const float* b1  = (const float*)d_in[6];
    const float* p1  = (const float*)d_in[7];
    const float* W2  = (const float*)d_in[8];
    const float* al2 = (const float*)d_in[9];
    const float* ar2 = (const float*)d_in[10];
    const float* b2  = (const float*)d_in[11];
    const float* p2  = (const float*)d_in[12];
    float* out = (float*)d_out;

    const int N = in_sizes[0] / NFEAT;
    const int E = in_sizes[1];
    const long long nf = (long long)N;

    // workspace layout (f32 region, then 16B-aligned f16 pool)
    float* ws   = (float*)d_ws;
    float* ftA  = ws;                      // N*512  (ft1 -> pf1; later ft2/pf2, stride NCLASS)
    float* aggB = ftA + nf * HID_ALL;      // N*512  (agg1; later agg2)
    float* el1  = aggB + nf * HID_ALL;     // N*8
    float* er1  = el1 + nf * NHEADS;       // N*8
    float* m1   = er1 + nf * NHEADS;       // N*8
    float* s1   = m1 + nf * NHEADS;        // N*8
    float* el2  = s1 + nf * NHEADS;        // N
    float* er2  = el2 + nf;                // N
    float* m2   = er2 + nf;                // N
    float* s2   = m2 + nf;                 // N
    float* mu   = s2 + nf;                 // 16 (mu1[0..7], mu2 at [8]; padded for alignment)
    half_t* xh  = (half_t*)(mu + 16);      // N*256 halves; reused as hcatH (N*512 halves)
    half_t* hcatH = xh;                    //   (xh dead after GEMM1)
    half_t* W1h = xh + nf * HID_ALL;       // 512*256 halves
    half_t* W2h = W1h + (long long)HID_ALL * NFEAT;  // 40*512 halves
    float* ft2  = ftA;                     // reuse
    float* agg2 = aggB;                    // reuse (zeroed after GEMM2 consumes hcat)

    const float INF = __builtin_inff();
    const int BS = 256;

    // ---- init + f16 conversions
    fill_f32<<<blocks_for(nf * HID_ALL, BS), BS, 0, stream>>>(aggB, 0.f, nf * HID_ALL);
    fill_f32<<<blocks_for(nf * NHEADS, BS), BS, 0, stream>>>(m1, -INF, nf * NHEADS);
    fill_f32<<<blocks_for(nf * NHEADS, BS), BS, 0, stream>>>(s1, 0.f, nf * NHEADS);
    fill_f32<<<1, 32, 0, stream>>>(mu, INF, 16);
    f32_to_f16<<<blocks_for(nf * NFEAT, BS), BS, 0, stream>>>(x, xh, nf * NFEAT);
    f32_to_f16<<<blocks_for((long long)HID_ALL * NFEAT, BS), BS, 0, stream>>>(W1, W1h, (long long)HID_ALL * NFEAT);
    f32_to_f16<<<blocks_for((long long)NCLASS * HID_ALL, BS), BS, 0, stream>>>(W2, W2h, (long long)NCLASS * HID_ALL);

    // ---- layer 1 GEMM: ft = x @ W1_all^T  (f16 WMMA + TDM-staged f16 B)
    {
        int rowTiles  = (N + 15) / 16;
        int rowGroups = (rowTiles + 7) / 8;
        int colGroups = (HID_ALL / 16 + 7) / 8;   // 32/8 = 4
        gemm_wmma_tdm<8><<<rowGroups * colGroups, BS, 0, stream>>>(xh, W1h, ftA, N, NFEAT, HID_ALL);
    }
    head_dots<<<blocks_for(nf * NHEADS, BS), BS, 0, stream>>>(ftA, al1, ar1, el1, er1, N, NHEADS, NHID);
    min_reduce<<<dim3(512, NHEADS), BS, 0, stream>>>(ftA, mu, N, HID_ALL, NHID);
    pow_fwd<<<blocks_for(nf * HID_ALL, BS), BS, 0, stream>>>(ftA, mu, p1, nf * HID_ALL, HID_ALL, NHID);

    edge_max_k<<<blocks_for(E, BS), BS, 0, stream>>>(src, dst, el1, er1, m1, E, NHEADS);
    edge_sum_k<<<blocks_for(E, BS), BS, 0, stream>>>(src, dst, el1, er1, m1, s1, E, NHEADS);
    edge_agg1<<<blocks_for((long long)E * NHEADS, BS), BS, 0, stream>>>(src, dst, el1, er1, m1, s1, ftA, aggB, E);
    // hcat (f16) written into half pool; xh is dead after GEMM1
    rst_elu1<<<blocks_for(nf * HID_ALL, BS), BS, 0, stream>>>(aggB, mu, p1, b1, hcatH, nf * HID_ALL);

    // ---- layer 2 GEMM: ft2 = hcat @ W2^T  (f16 WMMA + TDM, CT=3 covers 48 padded cols)
    {
        int rowTiles  = (N + 15) / 16;
        int rowGroups = (rowTiles + 7) / 8;
        gemm_wmma_tdm<3><<<rowGroups, BS, 0, stream>>>(hcatH, W2h, ft2, N, HID_ALL, NCLASS);
    }
    head_dots<<<blocks_for(nf, BS), BS, 0, stream>>>(ft2, al2, ar2, el2, er2, N, 1, NCLASS);
    min_reduce<<<dim3(512, 1), BS, 0, stream>>>(ft2, mu + 8, N, NCLASS, NCLASS);
    pow_fwd<<<blocks_for(nf * NCLASS, BS), BS, 0, stream>>>(ft2, mu + 8, p2, nf * NCLASS, NCLASS, NCLASS);

    // ---- init (layer 2) — after GEMM2 consumed hcat (stream-ordered)
    fill_f32<<<blocks_for(nf, BS), BS, 0, stream>>>(m2, -INF, nf);
    fill_f32<<<blocks_for(nf, BS), BS, 0, stream>>>(s2, 0.f, nf);
    fill_f32<<<blocks_for(nf * NCLASS, BS), BS, 0, stream>>>(agg2, 0.f, nf * NCLASS);

    edge_max_k<<<blocks_for(E, BS), BS, 0, stream>>>(src, dst, el2, er2, m2, E, 1);
    edge_sum_k<<<blocks_for(E, BS), BS, 0, stream>>>(src, dst, el2, er2, m2, s2, E, 1);
    edge_agg2<<<blocks_for(E, BS), BS, 0, stream>>>(src, dst, el2, er2, m2, s2, ft2, agg2, E);

    rst_logsoftmax<<<blocks_for(nf, BS), BS, 0, stream>>>(agg2, mu + 8, p2, b2, out, N);
}
